// CoevolutionHead_40785009443215
// MI455X (gfx1250) — matrix operand order repaired and checked by reading, not compiled, not run on hardware
//
#include <hip/hip_runtime.h>
#include <hip/hip_bf16.h>

typedef __attribute__((ext_vector_type(16))) _Float16 v16h;
typedef __attribute__((ext_vector_type(8)))  _Float16 v8h;
typedef __attribute__((ext_vector_type(8)))  float    v8f;

#define NN 256
#define MM 128
#define DSZ 384
#define DP 128
#define CC 22
#define C2 484                    // 22*22
#define NCT 31                    // ceil(484/16) col tiles
#define EIJ_ELEMS (NN*NN*C2)      // 31,719,424 floats
#define EI_OFF    EIJ_ELEMS
#define LOG_OFF   (EIJ_ELEMS + NN*CC)

// ---------------------------------------------------------------------------
// Kernel 0: convert + swizzle w3 [128,484] f32 -> f16 in WMMA B-fragment order.
// Layout: [ct (31)][kk (4)][lane (32)][e (16)] halves; each lane's 16 halves
// are contiguous (32B) so the hot loop loads one v16h per fragment.
// B 32x16 f16 layout: col = lane&15; K = kk*32 + (lane>=16 ? 8:0) + e + (e>=8 ? 8:0).
// Out-of-range cols zero-filled (no clamping needed in the GEMM).
// ---------------------------------------------------------------------------
__global__ void k_swz_w3(const float* __restrict__ w3, _Float16* __restrict__ w3s) {
    int idx = blockIdx.x * blockDim.x + threadIdx.x;   // 0 .. 31*4*32*16-1
    if (idx >= NCT*4*32*16) return;
    const int e    = idx & 15;
    const int lane = (idx >> 4) & 31;
    const int kk   = (idx >> 9) & 3;
    const int ct   = idx >> 11;
    const int col  = ct*16 + (lane & 15);
    const int K    = kk*32 + ((lane & 16) ? 8 : 0) + e + ((e & 8) ? 8 : 0);
    w3s[idx] = (col < C2) ? (_Float16)w3[(size_t)K*C2 + col] : (_Float16)0.f;
}

// ---------------------------------------------------------------------------
// Kernel 1: single branch. ei = LN(GELU(single@w1 + b1)) @ w2 + b2  [256,22]
// One block (128 threads) per sequence position n. ~75 MFLOP total: VALU is fine.
// ---------------------------------------------------------------------------
__global__ __launch_bounds__(128) void k_single(
    const float* __restrict__ single,
    const float* __restrict__ w1, const float* __restrict__ b1,
    const float* __restrict__ g1, const float* __restrict__ be1,
    const float* __restrict__ w2, const float* __restrict__ b2,
    float* __restrict__ ei) {
    __shared__ float x[DSZ];
    __shared__ float h[DSZ];
    __shared__ float rs[128], rq[128];
    const int n = blockIdx.x, t = threadIdx.x;

    for (int c = t; c < DSZ; c += 128) x[c] = single[n*DSZ + c];
    __syncthreads();

    float hv[3];
    float s = 0.f, q = 0.f;
    #pragma unroll
    for (int k = 0; k < 3; ++k) {
        const int c = t + k*128;
        float acc = b1[c];
        for (int d = 0; d < DSZ; ++d) acc = fmaf(x[d], w1[d*DSZ + c], acc);
        // exact GELU: 0.5*x*(1+erf(x/sqrt(2)))
        float g = 0.5f * acc * (1.0f + erff(acc * 0.70710678118654752f));
        hv[k] = g; s += g; q += g*g;
    }
    rs[t] = s; rq[t] = q;
    __syncthreads();
    for (int off = 64; off; off >>= 1) {
        if (t < off) { rs[t] += rs[t+off]; rq[t] += rq[t+off]; }
        __syncthreads();
    }
    const float mean = rs[0] * (1.0f/DSZ);
    const float var  = rq[0] * (1.0f/DSZ) - mean*mean;
    const float rstd = rsqrtf(var + 1e-5f);
    #pragma unroll
    for (int k = 0; k < 3; ++k) {
        const int c = t + k*128;
        h[c] = (hv[k] - mean) * rstd * g1[c] + be1[c];
    }
    __syncthreads();
    if (t < CC) {
        float acc = b2[t];
        for (int d = 0; d < DSZ; ++d) acc = fmaf(h[d], w2[d*CC + t], acc);
        ei[n*CC + t] = acc;
    }
}

// ---------------------------------------------------------------------------
// Kernel 2: pair branch (the big GEMM, WMMA f16 -> f32 accumulate).
// eij[i,j,:] = relu( LN(0.5*(pair[i,j]+pair[j,i])) @ w3 + b3 ), zero diagonal.
// Block = 256 threads (8 waves) handles rows (i, j0..j0+15) x all 484 cols.
// ---------------------------------------------------------------------------
#define AS 136  // LDS row stride in halves (multiple of 8 -> 16B-aligned chunks)
__global__ __launch_bounds__(256) void k_pair(
    const float* __restrict__ pair,
    const float* __restrict__ g2, const float* __restrict__ be2,
    const _Float16* __restrict__ w3s, const float* __restrict__ b3,
    float* __restrict__ eij) {
    __shared__ _Float16 A[16 * AS];
    const int i  = blockIdx.y;
    const int j0 = blockIdx.x * 16;
    const int t  = threadIdx.x;
    const int lane = t & 31;

    // --- load + symmetrize + LayerNorm + convert to f16 panel ---
    {
        const int r   = t >> 4;   // row within tile: 0..15
        const int sub = t & 15;   // 16 threads per row, 8 contiguous elems each
        const int j = j0 + r;
        float sv[8];
        float s = 0.f, q = 0.f;
        #pragma unroll
        for (int e = 0; e < 8; ++e) {
            const int d = sub*8 + e;
            const float a = pair[((size_t)i*NN + j)*DP + d];
            const float b = pair[((size_t)j*NN + i)*DP + d];
            const float v = 0.5f * (a + b);
            sv[e] = v; s += v; q += v*v;
        }
        #pragma unroll
        for (int off = 8; off; off >>= 1) {
            s += __shfl_xor(s, off, 16);
            q += __shfl_xor(q, off, 16);
        }
        const float mean = s * (1.0f/DP);
        const float var  = q * (1.0f/DP) - mean*mean;
        const float rstd = rsqrtf(var + 1e-5f);
        v8h pk;
        #pragma unroll
        for (int e = 0; e < 8; ++e) {
            const int d = sub*8 + e;
            pk[e] = (_Float16)((sv[e] - mean) * rstd * g2[d] + be2[d]);
        }
        *(v8h*)(&A[r*AS + sub*8]) = pk;   // one 16B LDS store
    }
    __syncthreads();

    // --- WMMA phase ---
    const int w = t >> 5;
    const int ar  = lane & 15;            // A row = lane&15
    const int kb8 = (lane >> 4) << 3;     // A: +8 K offset for upper half-wave
    // A fragments: 16-bit A 16x32 layout = two contiguous 8-half runs per lane.
    v16h af[4];
    #pragma unroll
    for (int kk = 0; kk < 4; ++kk) {
        const _Float16* ap = &A[ar*AS + kk*32 + kb8];
        v8h lo = *(const v8h*)(ap);        // K = kk*32+kb8 .. +7
        v8h hi = *(const v8h*)(ap + 16);   // K = kk*32+kb8+16 .. +23
        af[kk] = __builtin_shufflevector(lo, hi, 0,1,2,3,4,5,6,7,8,9,10,11,12,13,14,15);
    }

    const int colb  = lane & 15;
    const int rbase = (lane >> 4) << 3;   // D: M = v (+8 for upper half-wave)

    #pragma unroll
    for (int rep = 0; rep < 4; ++rep) {
        const int ct = w + rep*8;
        if (ct < NCT) {                   // wave-uniform guard: EXEC stays all-1s
            const _Float16* wp = w3s + ((size_t)ct*4)*512 + lane*16;
            v8f acc = {};
            #pragma unroll
            for (int kk = 0; kk < 4; ++kk) {
                v16h bf = *(const v16h*)(wp + kk*512);  // 32B contiguous per lane
                acc = __builtin_amdgcn_wmma_f32_16x16x32_f16(
                    false, af[kk], false, bf, (short)0, acc, false, false);
            }
            const int col = ct*16 + colb;
            if (col < C2) {
                const float bias = b3[col];
                #pragma unroll
                for (int v = 0; v < 8; ++v) {
                    const int j = j0 + v + rbase;
                    float f = acc[v] + bias;
                    f = f > 0.f ? f : 0.f;        // ReLU
                    if (j == i) f = 0.f;          // zero diagonal
                    eij[((size_t)i*NN + j)*C2 + col] = f;
                }
            }
        }
    }
}

// ---------------------------------------------------------------------------
// Kernel 3: coevolution gather. hi[m,i,c] = sum_j eij[i,j,c,msa[m,j]] (skip gap=21)
// One wave per (m,i). eij is L2-resident (127MB < 192MB L2).
// ---------------------------------------------------------------------------
__global__ __launch_bounds__(32) void k_logits(
    const float* __restrict__ eij, const float* __restrict__ ei,
    const int* __restrict__ msa, float* __restrict__ logits) {
    const int i = blockIdx.x;
    const int m = blockIdx.y;
    const int lane = threadIdx.x;
    float acc[CC];
    #pragma unroll
    for (int c = 0; c < CC; ++c) acc[c] = 0.f;

    for (int jj = 0; jj < NN; jj += 32) {
        const int j = jj + lane;
        const int d = msa[m*NN + j];
        if (d < CC - 1) {   // gap class (21) is masked out
            const float* p = eij + ((size_t)i*NN + j)*C2 + d;
            #pragma unroll
            for (int c = 0; c < CC; ++c) acc[c] += p[c*CC];
        }
    }
    #pragma unroll
    for (int c = 0; c < CC; ++c) {
        #pragma unroll
        for (int off = 16; off; off >>= 1) acc[c] += __shfl_xor(acc[c], off, 32);
    }
    if (lane == 0) {
        #pragma unroll
        for (int c = 0; c < CC; ++c)
            logits[((size_t)m*NN + i)*CC + c] = ei[i*CC + c] + acc[c];
    }
}

// ---------------------------------------------------------------------------
extern "C" void kernel_launch(void* const* d_in, const int* in_sizes, int n_in,
                              void* d_out, int out_size, void* d_ws, size_t ws_size,
                              hipStream_t stream) {
    (void)in_sizes; (void)n_in; (void)out_size; (void)ws_size;
    const float* single = (const float*)d_in[0];
    const float* pair   = (const float*)d_in[1];
    const int*   msa    = (const int*)  d_in[2];
    const float* w1  = (const float*)d_in[3];
    const float* b1  = (const float*)d_in[4];
    const float* g1  = (const float*)d_in[5];
    const float* be1 = (const float*)d_in[6];
    const float* w2  = (const float*)d_in[7];
    const float* b2  = (const float*)d_in[8];
    const float* g2  = (const float*)d_in[9];
    const float* be2 = (const float*)d_in[10];
    const float* w3  = (const float*)d_in[11];
    const float* b3  = (const float*)d_in[12];

    float* out    = (float*)d_out;
    float* eij    = out;
    float* ei     = out + EI_OFF;
    float* logits = out + LOG_OFF;
    _Float16* w3s = (_Float16*)d_ws;      // 31*4*32*16 halves = 124 KB

    const int nswz = NCT*4*32*16;
    k_swz_w3 <<<(nswz + 255)/256, 256, 0, stream>>>(w3, w3s);
    k_single <<<NN, 128, 0, stream>>>(single, w1, b1, g1, be1, w2, b2, ei);
    k_pair   <<<dim3(16, NN), 256, 0, stream>>>(pair, g2, be2, w3s, b3, eij);
    k_logits <<<dim3(NN, MM), 32, 0, stream>>>(eij, ei, msa, logits);
}
